// UNet_11828339933632
// MI455X (gfx1250) — matrix operand order repaired and checked by reading, not compiled, not run on hardware
//
#include <hip/hip_runtime.h>
#include <cstddef>

// ---------------------------------------------------------------------------
// Types for CDNA5 WMMA (wave32). bf16 activations/weights, f32 accumulate.
// ---------------------------------------------------------------------------
typedef __bf16 bf16;
typedef __attribute__((ext_vector_type(16))) __bf16        v16bf;
typedef __attribute__((ext_vector_type(8)))  float         v8f;
typedef __attribute__((ext_vector_type(4)))  unsigned int  v4u;

union ABu { v16bf v; v4u q[2]; };

// ---------------------------------------------------------------------------
// Gathered GEMM:  Y[n,co] = sum_k sum_ci X[row(n,k), ci] * Wpk[k,ci,co]
//   row(n,k) = idx[n*K+k] (HASIDX) or n (identity).  Edge rows clamp to 0 and
//   are store-guarded.  Ci = 32*NB (compile-time), Co = 16*ct.
// One wave computes a 16 x (16*CTB) strip: each gathered A fragment is reused
// across CTB independent WMMAs (register blocking over N) -> CTB x less
// gather traffic and CTB independent v_wmma per A load for latency hiding.
// 4 waves/block share the same co-tile group (B hits L2/L0).
// A layout per ISA (16-bit A 16x32): lanes 0-15 row M=lane, VGPR0-3 K=0..7,
// VGPR4-7 K=16..23; lanes 16-31 same rows, K offset +8 -> xq[hi], xq[hi+2].
// B pre-packed: 32 contiguous bytes per lane per tile; adjacent co-tiles are
// 1KB apart, so the CTB B loads are sequential b128 pairs.
// ---------------------------------------------------------------------------
template <int NB, int CTB, bool HASIDX>
__global__ __launch_bounds__(128)
void k_gemm(const bf16* __restrict__ X, int ldx,
            const int* __restrict__ idx, int K,
            const bf16* __restrict__ Wpk, int ct,
            bf16* __restrict__ Y, int ldy, int N)
{
  const int lane = threadIdx.x & 31;
  const int wid  = threadIdx.x >> 5;
  const int m    = lane & 15;
  const int hi   = lane >> 4;
  const int tm   = blockIdx.x * 4 + wid;
  const int tn0  = blockIdx.y * CTB;
  const int row  = tm * 16 + m;
  const int rowc = (row < N) ? row : 0;   // clamp: OOB rows compute garbage, never stored

  const int* ip = HASIDX ? (idx + (size_t)rowc * K) : nullptr;
  const bf16* bbase = Wpk + (size_t)tn0 * 512 + lane * 16;
  const size_t bstep = (size_t)ct * 512;   // bf16 elems between consecutive (kk,cb) B tiles

  v8f acc[CTB];
  v8f zero = {};
#pragma unroll
  for (int j = 0; j < CTB; ++j) acc[j] = zero;

  for (int kk = 0; kk < K; ++kk) {
    const int r = HASIDX ? ip[kk] : rowc;
    const bf16* xrow = X + (unsigned)r * (unsigned)ldx;
    const bf16* bp = bbase + (size_t)(kk * NB) * bstep;
#pragma unroll
    for (int cb = 0; cb < NB; ++cb) {
      ABu a;
      const v4u* xq = (const v4u*)(xrow + cb * 32);
      a.q[0] = xq[hi];
      a.q[1] = xq[hi + 2];
      ABu b[CTB];
      const bf16* bcb = bp + (size_t)cb * bstep;
#pragma unroll
      for (int j = 0; j < CTB; ++j) {
        const v4u* bq = (const v4u*)(bcb + j * 512);
        b[j].q[0] = bq[0];
        b[j].q[1] = bq[1];
      }
#pragma unroll
      for (int j = 0; j < CTB; ++j)
        acc[j] = __builtin_amdgcn_wmma_f32_16x16x32_bf16(false, a.v, false, b[j].v,
                                                         (short)0, acc[j], false, false);
    }
  }

  const int r0 = tm * 16 + hi * 8;
  const bool full = (r0 + 8 <= N);
#pragma unroll
  for (int j = 0; j < CTB; ++j) {
    const int col = (tn0 + j) * 16 + m;
    unsigned ybase = (unsigned)r0 * (unsigned)ldy + (unsigned)col;
    if (full) {
#pragma unroll
      for (int i = 0; i < 8; ++i)
        Y[ybase + (unsigned)(i * ldy)] = (bf16)acc[j][i];
    } else {
#pragma unroll
      for (int i = 0; i < 8; ++i)
        if (r0 + i < N) Y[ybase + (unsigned)(i * ldy)] = (bf16)acc[j][i];
    }
  }
}

// Pack W [K,CiReal,Co] (f32) -> WMMA-B layout bf16:
//   P[((kk*nb+cb)*ct+tn)*512 + lane*16 + e] = W[kk][cb*32+(e<8 ? 8*hi+e : 16+8*hi+e-8)][tn*16+(lane&15)]
// zero-padded where ci >= CiReal.
__global__ void k_pack(const float* __restrict__ W, bf16* __restrict__ P,
                       int K, int CiReal, int nb, int ct, int Co)
{
  size_t t = (size_t)blockIdx.x * 256 + threadIdx.x;
  size_t total = (size_t)K * nb * ct * 512;
  if (t >= total) return;
  int e    = (int)(t & 15);
  int lane = (int)((t >> 4) & 31);
  size_t rest = t >> 9;
  int tn = (int)(rest % ct); rest /= ct;
  int cb = (int)(rest % nb);
  int kk = (int)(rest / nb);
  int hi = lane >> 4, n = lane & 15;
  int ci = cb * 32 + (e < 8 ? 8 * hi + e : 16 + 8 * hi + (e - 8));
  int co = tn * 16 + n;
  float v = (ci < CiReal) ? W[((size_t)kk * CiReal + ci) * Co + co] : 0.0f;
  P[t] = (bf16)v;
}

// GroupNorm(groups of 4) + optional second-GN shortcut (S,gs,bs)
// + optional bf16 residual R + ReLU.  One thread per (node, group-of-4).
__global__ void k_gn(const bf16* __restrict__ T,
                     const float* __restrict__ g, const float* __restrict__ b,
                     const bf16* S, const float* gs, const float* bs,
                     const bf16* R, int ldr,
                     bf16* D, int ldd, int doff,
                     int N, int C, int relu)
{
  int t = blockIdx.x * 256 + threadIdx.x;
  int ng = C >> 2;
  if (t >= N * ng) return;
  int n = t / ng;
  int c0 = (t - n * ng) * 4;
  const bf16* tp = T + (size_t)n * C + c0;
  float x0 = (float)tp[0], x1 = (float)tp[1], x2 = (float)tp[2], x3 = (float)tp[3];
  float mu = 0.25f * (x0 + x1 + x2 + x3);
  float d0 = x0 - mu, d1 = x1 - mu, d2 = x2 - mu, d3 = x3 - mu;
  float inv = rsqrtf(0.25f * (d0*d0 + d1*d1 + d2*d2 + d3*d3) + 1e-5f);
  float y0 = d0 * inv * g[c0+0] + b[c0+0];
  float y1 = d1 * inv * g[c0+1] + b[c0+1];
  float y2 = d2 * inv * g[c0+2] + b[c0+2];
  float y3 = d3 * inv * g[c0+3] + b[c0+3];
  if (S) {
    const bf16* sp = S + (size_t)n * C + c0;
    float s0 = (float)sp[0], s1 = (float)sp[1], s2 = (float)sp[2], s3 = (float)sp[3];
    float smu = 0.25f * (s0 + s1 + s2 + s3);
    float e0 = s0 - smu, e1 = s1 - smu, e2 = s2 - smu, e3 = s3 - smu;
    float sinv = rsqrtf(0.25f * (e0*e0 + e1*e1 + e2*e2 + e3*e3) + 1e-5f);
    y0 += e0 * sinv * gs[c0+0] + bs[c0+0];
    y1 += e1 * sinv * gs[c0+1] + bs[c0+1];
    y2 += e2 * sinv * gs[c0+2] + bs[c0+2];
    y3 += e3 * sinv * gs[c0+3] + bs[c0+3];
  }
  if (R) {
    const bf16* rp = R + (size_t)n * ldr + c0;
    y0 += (float)rp[0]; y1 += (float)rp[1]; y2 += (float)rp[2]; y3 += (float)rp[3];
  }
  if (relu) {
    y0 = fmaxf(y0, 0.0f); y1 = fmaxf(y1, 0.0f);
    y2 = fmaxf(y2, 0.0f); y3 = fmaxf(y3, 0.0f);
  }
  bf16* dp = D + (size_t)n * ldd + doff + c0;
  dp[0] = (bf16)y0; dp[1] = (bf16)y1; dp[2] = (bf16)y2; dp[3] = (bf16)y3;
}

// Deconv -> gather-conv index: idx2[n][o] = (oct[n]==o) ? parent[n] : zrow
__global__ void k_didx(const int* __restrict__ par, const int* __restrict__ oct,
                       int* __restrict__ out, int N, int zrow)
{
  int n = blockIdx.x * 256 + threadIdx.x;
  if (n >= N) return;
  int p = par[n], o = oct[n];
#pragma unroll
  for (int k = 0; k < 8; ++k) out[n * 8 + k] = (k == o) ? p : zrow;
}

__global__ void k_zero(bf16* p, int n)
{
  int t = blockIdx.x * 256 + threadIdx.x;
  if (t < n) p[t] = (bf16)0.0f;
}

// data [N,4] f32 -> X [(N+1),32] bf16, zero-padded cols and zero row N
__global__ void k_cvt_in(const float* __restrict__ d, bf16* __restrict__ X, int N)
{
  int t = blockIdx.x * 256 + threadIdx.x;
  if (t >= (N + 1) * 32) return;
  int n = t >> 5, c = t & 31;
  float v = (n < N && c < 4) ? d[n * 4 + c] : 0.0f;
  X[t] = (bf16)v;
}

// out[q,c] = sum_k qw[q,k] * D8[qidx[q,k], c]   (C=96)
__global__ void k_query(const float* __restrict__ qw, const int* __restrict__ qidx,
                        const bf16* __restrict__ D8, float* __restrict__ out, int Q)
{
  int t = blockIdx.x * 256 + threadIdx.x;
  if (t >= Q * 96) return;
  int q = t / 96, c = t - q * 96;
  float s = 0.0f;
#pragma unroll
  for (int k = 0; k < 8; ++k)
    s += qw[q * 8 + k] * (float)D8[(size_t)qidx[q * 8 + k] * 96 + c];
  out[t] = s;
}

// ---------------------------------------------------------------------------
// Host orchestration
// ---------------------------------------------------------------------------
struct RBp {
  const float *w1, *g1, *b1, *w2, *g2, *b2, *wsc, *gsc, *bsc;
};

extern "C" void kernel_launch(void* const* d_in, const int* in_sizes, int n_in,
                              void* d_out, int out_size, void* d_ws, size_t ws_size,
                              hipStream_t stream)
{
  (void)in_sizes; (void)out_size; (void)ws_size;
  if (n_in < 22) return;
  const int N8 = 100000, N7 = 40000, N6 = 16000, N5 = 6400, N4 = 2560, Q = 50000;

  // ---- walk params pytree (jax sorted-dict leaf order) --------------------
  const bool pk = (n_in <= 40);          // params passed as one flat buffer
  const float* pbase = (const float*)d_in[2];
  size_t poff = 0;
  int leafIdx = 0;
  auto next = [&](size_t sz) -> const float* {
    if (pk) { const float* p = pbase + poff; poff += sz; return p; }
    return (const float*)d_in[2 + leafIdx++];
  };

  // conv1: b,g,w
  const float* c1b = next(32);
  const float* c1g = next(32);
  const float* c1w = next((size_t)27 * 4 * 32);
  // dec (4 stages x [block0 w/ wsc, block1])
  static const int DCo[4]  = {256, 128, 96, 96};
  static const int DCi0[4] = {384, 192, 128, 128};
  RBp dec[4][2];
  for (int i = 0; i < 4; ++i) {
    int co = DCo[i], ci0 = DCi0[i];
    RBp r = {};
    r.b1 = next(co); r.b2 = next(co); r.bsc = next(co);
    r.g1 = next(co); r.g2 = next(co); r.gsc = next(co);
    r.w1 = next((size_t)27 * ci0 * co);
    r.w2 = next((size_t)27 * co * co);
    r.wsc = next((size_t)ci0 * co);
    dec[i][0] = r;
    RBp s = {};
    s.b1 = next(co); s.b2 = next(co); s.g1 = next(co); s.g2 = next(co);
    s.w1 = next((size_t)27 * co * co);
    s.w2 = next((size_t)27 * co * co);
    dec[i][1] = s;
  }
  // down
  static const int dnci[4] = {32, 32, 64, 128}, dnco[4] = {32, 64, 128, 256};
  const float *dnb[4], *dng[4], *dnw[4];
  for (int i = 0; i < 4; ++i) {
    dnb[i] = next(dnco[i]); dng[i] = next(dnco[i]);
    dnw[i] = next((size_t)8 * dnci[i] * dnco[i]);
  }
  // enc
  static const int encC[4] = {32, 64, 128, 256}, encB[4] = {2, 3, 4, 6};
  RBp enc[4][6];
  for (int i = 0; i < 4; ++i)
    for (int j = 0; j < encB[i]; ++j) {
      int C = encC[i];
      RBp r = {};
      r.b1 = next(C); r.b2 = next(C); r.g1 = next(C); r.g2 = next(C);
      r.w1 = next((size_t)27 * C * C);
      r.w2 = next((size_t)27 * C * C);
      enc[i][j] = r;
    }
  // up
  static const int upci[4] = {256, 256, 128, 96}, upco[4] = {256, 128, 96, 96};
  const float *ub[4], *ug[4], *uw[4];
  for (int i = 0; i < 4; ++i) {
    ub[i] = next(upco[i]); ug[i] = next(upco[i]);
    uw[i] = next((size_t)8 * upci[i] * upco[i]);
  }

  const int tb = pk ? 3 : (2 + leafIdx);
  const float* data  = (const float*)d_in[0];
  const float* qw    = (const float*)d_in[1];
  const int* neigh8  = (const int*)d_in[tb + 0];
  const int* neigh7  = (const int*)d_in[tb + 1];
  const int* neigh6  = (const int*)d_in[tb + 2];
  const int* neigh5  = (const int*)d_in[tb + 3];
  const int* neigh4  = (const int*)d_in[tb + 4];
  const int* down7   = (const int*)d_in[tb + 5];
  const int* down6   = (const int*)d_in[tb + 6];
  const int* down5   = (const int*)d_in[tb + 7];
  const int* down4   = (const int*)d_in[tb + 8];
  const int* upar[4] = {(const int*)d_in[tb + 9],  (const int*)d_in[tb + 11],
                        (const int*)d_in[tb + 13], (const int*)d_in[tb + 15]};
  const int* uoct[4] = {(const int*)d_in[tb + 10], (const int*)d_in[tb + 12],
                        (const int*)d_in[tb + 14], (const int*)d_in[tb + 16]};
  const int* qidx    = (const int*)d_in[tb + 17];

  // ---- workspace carve ----------------------------------------------------
  size_t off = 0;
  auto carve = [&](size_t bytes) -> void* {
    void* p = (char*)d_ws + off;
    off += (bytes + 255) & ~(size_t)255;
    return p;
  };
  bf16* X8in = (bf16*)carve((size_t)(N8 + 1) * 32 * 2);
  bf16* C8   = (bf16*)carve((size_t)(N8 + 1) * 128 * 2);  // [conv1 32 | up8 96]
  bf16* C7   = (bf16*)carve((size_t)(N7 + 1) * 128 * 2);  // [enc7 32 | up7 96]
  bf16* C6   = (bf16*)carve((size_t)(N6 + 1) * 192 * 2);  // [enc6 64 | up6 128]
  bf16* C5   = (bf16*)carve((size_t)(N5 + 1) * 384 * 2);  // [enc5 128| up5 256]
  bf16* X4   = (bf16*)carve((size_t)(N4 + 1) * 256 * 2);
  bf16* D5   = (bf16*)carve((size_t)(N5 + 1) * 256 * 2);
  bf16* D6   = (bf16*)carve((size_t)(N6 + 1) * 128 * 2);
  bf16* D7   = (bf16*)carve((size_t)(N7 + 1) * 96 * 2);
  bf16* D8   = (bf16*)carve((size_t)(N8 + 1) * 96 * 2);
  bf16* ABUF = (bf16*)carve((size_t)N8 * 96 * 2);
  bf16* TMP  = (bf16*)carve((size_t)N8 * 96 * 2);
  bf16* SC   = (bf16*)carve((size_t)N8 * 96 * 2);
  bf16* WPK  = (bf16*)carve((size_t)27 * 384 * 256 * 2);
  int*  IDX2 = (int*)carve((size_t)N8 * 8 * 4);

  // ---- helpers ------------------------------------------------------------
  auto conv = [&](const bf16* X, int ldx, const int* idx, int K,
                  const float* W, int CiReal, int CiPad, int Co, int N, bf16* T) {
    int nb = CiPad / 32, ct = Co / 16;
    int ctb = (Co == 32) ? 2 : (Co == 96 ? 3 : 4);
    size_t pe = (size_t)K * nb * ct * 512;
    k_pack<<<dim3((unsigned)((pe + 255) / 256)), dim3(256), 0, stream>>>(
        W, WPK, K, CiReal, nb, ct, Co);
    dim3 g((N + 63) / 64, ct / ctb), blk(128);
#define GEMM_CASE(NBv, CTBv)                                                   \
    else if (nb == NBv && ctb == CTBv) {                                       \
      if (idx) k_gemm<NBv, CTBv, true ><<<g, blk, 0, stream>>>(X, ldx, idx, K, \
                                                      WPK, ct, T, Co, N);      \
      else     k_gemm<NBv, CTBv, false><<<g, blk, 0, stream>>>(X, ldx, idx, K, \
                                                      WPK, ct, T, Co, N);      \
    }
    if (false) {}
    GEMM_CASE(1, 2)  GEMM_CASE(1, 4)  GEMM_CASE(2, 4)
    GEMM_CASE(3, 3)  GEMM_CASE(4, 3)  GEMM_CASE(4, 4)
    GEMM_CASE(6, 4)  GEMM_CASE(8, 4)  GEMM_CASE(12, 4)
#undef GEMM_CASE
  };
  auto gnorm = [&](const bf16* T, const float* g, const float* b,
                   const bf16* S, const float* gs, const float* bs,
                   const bf16* R, int ldr,
                   bf16* D, int ldd, int doff, int N, int C) {
    int tot = N * (C / 4);
    k_gn<<<dim3((tot + 255) / 256), dim3(256), 0, stream>>>(
        T, g, b, S, gs, bs, R, ldr, D, ldd, doff, N, C, 1);
  };
  auto resblock = [&](bf16* X, int ldx, int Cin, int C, const int* neigh, int N,
                      const RBp& p, bf16* OUT, int ldo) {
    conv(X, ldx, neigh, 27, p.w1, Cin, Cin, C, N, TMP);
    gnorm(TMP, p.g1, p.b1, nullptr, nullptr, nullptr, nullptr, 0, ABUF, C, 0, N, C);
    conv(ABUF, C, neigh, 27, p.w2, C, C, C, N, TMP);
    if (p.wsc) {
      conv(X, ldx, nullptr, 1, p.wsc, Cin, Cin, C, N, SC);
      gnorm(TMP, p.g2, p.b2, SC, p.gsc, p.bsc, nullptr, 0, OUT, ldo, 0, N, C);
    } else {
      gnorm(TMP, p.g2, p.b2, nullptr, nullptr, nullptr, X, ldx, OUT, ldo, 0, N, C);
    }
  };

  // ---- pipeline -----------------------------------------------------------
  k_cvt_in<<<dim3(((N8 + 1) * 32 + 255) / 256), dim3(256), 0, stream>>>(data, X8in, N8);
  k_zero<<<1, 256, 0, stream>>>(X4 + (size_t)N4 * 256, 256);
  k_zero<<<1, 256, 0, stream>>>(D5 + (size_t)N5 * 256, 256);
  k_zero<<<1, 256, 0, stream>>>(D6 + (size_t)N6 * 128, 128);
  k_zero<<<1, 256, 0, stream>>>(D7 + (size_t)N7 * 96, 96);

  // conv1 -> C8[:, 0:32]
  conv(X8in, 32, neigh8, 27, c1w, 4, 32, 32, N8, TMP);
  gnorm(TMP, c1g, c1b, nullptr, nullptr, nullptr, nullptr, 0, C8, 128, 0, N8, 32);

  // encoder
  conv(C8, 128, down7, 8, dnw[0], 32, 32, 32, N7, TMP);
  gnorm(TMP, dng[0], dnb[0], nullptr, nullptr, nullptr, nullptr, 0, C7, 128, 0, N7, 32);
  for (int j = 0; j < 2; ++j) resblock(C7, 128, 32, 32, neigh7, N7, enc[0][j], C7, 128);

  conv(C7, 128, down6, 8, dnw[1], 32, 32, 64, N6, TMP);
  gnorm(TMP, dng[1], dnb[1], nullptr, nullptr, nullptr, nullptr, 0, C6, 192, 0, N6, 64);
  for (int j = 0; j < 3; ++j) resblock(C6, 192, 64, 64, neigh6, N6, enc[1][j], C6, 192);

  conv(C6, 192, down5, 8, dnw[2], 64, 64, 128, N5, TMP);
  gnorm(TMP, dng[2], dnb[2], nullptr, nullptr, nullptr, nullptr, 0, C5, 384, 0, N5, 128);
  for (int j = 0; j < 4; ++j) resblock(C5, 384, 128, 128, neigh5, N5, enc[2][j], C5, 384);

  conv(C5, 384, down4, 8, dnw[3], 128, 128, 256, N4, TMP);
  gnorm(TMP, dng[3], dnb[3], nullptr, nullptr, nullptr, nullptr, 0, X4, 256, 0, N4, 256);
  for (int j = 0; j < 6; ++j) resblock(X4, 256, 256, 256, neigh4, N4, enc[3][j], X4, 256);

  // decoder
  k_didx<<<dim3((N5 + 255) / 256), dim3(256), 0, stream>>>(upar[0], uoct[0], IDX2, N5, N4);
  conv(X4, 256, IDX2, 8, uw[0], 256, 256, 256, N5, TMP);
  gnorm(TMP, ug[0], ub[0], nullptr, nullptr, nullptr, nullptr, 0, C5, 384, 128, N5, 256);
  resblock(C5, 384, 384, 256, neigh5, N5, dec[0][0], D5, 256);
  resblock(D5, 256, 256, 256, neigh5, N5, dec[0][1], D5, 256);

  k_didx<<<dim3((N6 + 255) / 256), dim3(256), 0, stream>>>(upar[1], uoct[1], IDX2, N6, N5);
  conv(D5, 256, IDX2, 8, uw[1], 256, 256, 128, N6, TMP);
  gnorm(TMP, ug[1], ub[1], nullptr, nullptr, nullptr, nullptr, 0, C6, 192, 64, N6, 128);
  resblock(C6, 192, 192, 128, neigh6, N6, dec[1][0], D6, 128);
  resblock(D6, 128, 128, 128, neigh6, N6, dec[1][1], D6, 128);

  k_didx<<<dim3((N7 + 255) / 256), dim3(256), 0, stream>>>(upar[2], uoct[2], IDX2, N7, N6);
  conv(D6, 128, IDX2, 8, uw[2], 128, 128, 96, N7, TMP);
  gnorm(TMP, ug[2], ub[2], nullptr, nullptr, nullptr, nullptr, 0, C7, 128, 32, N7, 96);
  resblock(C7, 128, 128, 96, neigh7, N7, dec[2][0], D7, 96);
  resblock(D7, 96, 96, 96, neigh7, N7, dec[2][1], D7, 96);

  k_didx<<<dim3((N8 + 255) / 256), dim3(256), 0, stream>>>(upar[3], uoct[3], IDX2, N8, N7);
  conv(D7, 96, IDX2, 8, uw[3], 96, 96, 96, N8, TMP);
  gnorm(TMP, ug[3], ub[3], nullptr, nullptr, nullptr, nullptr, 0, C8, 128, 32, N8, 96);
  resblock(C8, 128, 128, 96, neigh8, N8, dec[3][0], D8, 96);
  resblock(D8, 96, 96, 96, neigh8, N8, dec[3][1], D8, 96);

  // query interpolation
  k_query<<<dim3((Q * 96 + 255) / 256), dim3(256), 0, stream>>>(
      qw, qidx, D8, (float*)d_out, Q);
}
